// MambaAudioModel_63221918597211
// MI455X (gfx1250) — matrix operand order repaired and checked by reading, not compiled, not run on hardware
//
#include <hip/hip_runtime.h>
#include <hip/hip_bf16.h>
#include <stdint.h>

// ---------------------------------------------------------------------------
// Model dimensions (fixed by the reference)
// ---------------------------------------------------------------------------
static constexpr int Vv  = 1024;
static constexpr int Dd  = 512;
static constexpr int Tt  = 2048;
static constexpr int Bb  = 2;
static constexpr int Ll  = 4;
static constexpr int DSs = 16;
static constexpr int DCc = 4;
static constexpr int DIi = 1024;           // E*D
static constexpr int Rr  = 32;             // (D+15)//16
static constexpr int Mm  = Bb * Tt;        // 4096 rows in all GEMMs

typedef unsigned short u16;
typedef __attribute__((ext_vector_type(16))) __bf16 bf16x16;
typedef __attribute__((ext_vector_type(8)))  float  v8f;

struct __align__(16) U4 { uint32_t x, y, z, w; };
union FragBF { bf16x16 v; U4 q[2]; };

__device__ __forceinline__ u16 f2bf(float f) {
  uint32_t u = __float_as_uint(f);
  u += 0x7FFFu + ((u >> 16) & 1u);       // round-to-nearest-even
  return (u16)(u >> 16);
}
__device__ __forceinline__ float sigmoidf_(float x) { return 1.f / (1.f + __expf(-x)); }

// ---------------------------------------------------------------------------
// Embedding: x[b,t,:] = tok_emb[ids[b,t]] + pos_emb[t]
// ---------------------------------------------------------------------------
__global__ void embed_k(const int* __restrict__ ids, const float* __restrict__ tok,
                        const float* __restrict__ pos, float* __restrict__ x, size_t total) {
  size_t i = (size_t)blockIdx.x * blockDim.x + threadIdx.x;
  if (i >= total) return;
  int d = (int)(i % Dd);
  size_t m = i / Dd;
  int t = (int)(m % Tt);
  x[i] = tok[(size_t)ids[m] * Dd + d] + pos[(size_t)t * Dd + d];
}

// ---------------------------------------------------------------------------
// f32 -> bf16 weight cast
// ---------------------------------------------------------------------------
__global__ void castw_k(const float* __restrict__ src, u16* __restrict__ dst, size_t n) {
  size_t i = (size_t)blockIdx.x * blockDim.x + threadIdx.x;
  if (i < n) dst[i] = f2bf(src[i]);
}

// f32 (strided rows) -> bf16 (dense), optional ReLU
__global__ void cast_rows_k(const float* __restrict__ src, int src_stride,
                            u16* __restrict__ dst, int cols, int relu, size_t total) {
  size_t i = (size_t)blockIdx.x * blockDim.x + threadIdx.x;
  if (i >= total) return;
  int c = (int)(i % cols);
  size_t r = i / cols;
  float v = src[r * (size_t)src_stride + c];
  if (relu) v = fmaxf(v, 0.f);
  dst[i] = f2bf(v);
}

// x += t
__global__ void add_k(float* __restrict__ x, const float* __restrict__ t, size_t total) {
  size_t i = (size_t)blockIdx.x * blockDim.x + threadIdx.x;
  if (i < total) x[i] += t[i];
}

// ---------------------------------------------------------------------------
// LayerNorm (row of D=512) + bf16 cast.  One wave32 per row, shuffle reduce.
// ---------------------------------------------------------------------------
__global__ __launch_bounds__(256) void ln_cast_k(const float* __restrict__ x,
                                                 const float* __restrict__ g,
                                                 const float* __restrict__ b,
                                                 u16* __restrict__ out) {
  int row  = blockIdx.x * 8 + (threadIdx.x >> 5);
  int lane = threadIdx.x & 31;
  const float* xr = x + (size_t)row * Dd;
  float s = 0.f, s2 = 0.f;
  for (int d = lane; d < Dd; d += 32) { float v = xr[d]; s += v; s2 += v * v; }
#pragma unroll
  for (int off = 16; off > 0; off >>= 1) {
    s  += __shfl_xor(s,  off, 32);
    s2 += __shfl_xor(s2, off, 32);
  }
  float mean = s * (1.f / Dd);
  float var  = s2 * (1.f / Dd) - mean * mean;
  float inv  = rsqrtf(var + 1e-5f);
  u16* orow = out + (size_t)row * Dd;
  for (int d = lane; d < Dd; d += 32)
    orow[d] = f2bf((xr[d] - mean) * inv * g[d] + b[d]);
}

// ---------------------------------------------------------------------------
// WMMA bf16 GEMM:  C[M,N] = A[M,K] * W[N,K]^T (+ bias[n])
// A, W row-major bf16; C row-major f32.
// Wave computes a 32(M) x 64(N) strip: 2 A frags x 4 B frags -> 8 WMMAs/K-step.
// Fragment sets are double-buffered so loads for step k+32 are in flight while
// the WMMAs for step k issue (kills the s_wait_loadcnt 0 before each WMMA).
// Block = 8 waves -> 256 x 64 tile.  Layouts per CDNA5 ISA 7.12.2.
// ---------------------------------------------------------------------------
struct FragSet { FragBF a[2]; FragBF b[4]; };

__device__ __forceinline__ void load_frags(FragSet& f, const u16* pa0, const u16* pa1,
                                           const u16* const* pw, int k) {
  f.a[0].q[0] = *(const U4*)(pa0 + k);
  f.a[0].q[1] = *(const U4*)(pa0 + k + 16);
  f.a[1].q[0] = *(const U4*)(pa1 + k);
  f.a[1].q[1] = *(const U4*)(pa1 + k + 16);
#pragma unroll
  for (int j = 0; j < 4; ++j) {
    f.b[j].q[0] = *(const U4*)(pw[j] + k);
    f.b[j].q[1] = *(const U4*)(pw[j] + k + 16);
  }
}

__device__ __forceinline__ void do_wmma(v8f acc[2][4], const FragSet& f) {
#pragma unroll
  for (int i = 0; i < 2; ++i)
#pragma unroll
    for (int j = 0; j < 4; ++j)
      acc[i][j] = __builtin_amdgcn_wmma_f32_16x16x32_bf16(
          false, f.a[i].v, false, f.b[j].v, (short)0, acc[i][j], false, false);
}

__global__ __launch_bounds__(256) void gemm_bf16_wmma(
    const u16* __restrict__ A, const u16* __restrict__ W,
    float* __restrict__ C, const float* __restrict__ bias,
    int N, int K) {
  const int lane = threadIdx.x & 31;
  const int wave = threadIdx.x >> 5;
  const int m0   = (blockIdx.y * 8 + wave) * 32;
  const int n0   = blockIdx.x * 64;
  const int row  = lane & 15;
  const int hi   = lane >> 4;

  const u16* pa0 = A + (size_t)(m0 + row) * K + hi * 8;   // A: lane holds K octets
  const u16* pa1 = pa0 + (size_t)16 * K;
  const u16* pw[4];
#pragma unroll
  for (int j = 0; j < 4; ++j)                             // B: lane holds K octets
    pw[j] = W + (size_t)(n0 + j * 16 + row) * K + hi * 8;

  v8f acc[2][4] = {};
  FragSet f0, f1;
  load_frags(f0, pa0, pa1, pw, 0);

  if (K > 32) {                 // all such K are multiples of 64
    for (int k = 0; k < K; k += 64) {
      __builtin_prefetch(pa0 + k + 1024, 0, 3);
      __builtin_prefetch(pa1 + k + 1024, 0, 3);
      load_frags(f1, pa0, pa1, pw, k + 32);   // in flight during f0 WMMAs
      do_wmma(acc, f0);
      if (k + 64 < K) load_frags(f0, pa0, pa1, pw, k + 64);
      do_wmma(acc, f1);
    }
  } else {                      // K == 32 (dt GEMM): single step
    do_wmma(acc, f0);
  }

#pragma unroll
  for (int i = 0; i < 2; ++i) {
#pragma unroll
    for (int j = 0; j < 4; ++j) {
      const int col = n0 + j * 16 + row;
      const float bv = bias ? bias[col] : 0.f;
      float* cp = C + (size_t)(m0 + i * 16 + hi * 8) * N + col;
#pragma unroll
      for (int v = 0; v < 8; ++v)
        cp[(size_t)v * N] = acc[i][j][v] + bv;  // lane holds col; rows hi*8+v
    }
  }
}

// ---------------------------------------------------------------------------
// Depthwise causal conv (k=4) + bias + SiLU; emits f32 and bf16 copies.
// xz layout [M, 2*DI]; xs = first DI columns.
// ---------------------------------------------------------------------------
__global__ void conv_silu_k(const float* __restrict__ xz, const float* __restrict__ cw,
                            const float* __restrict__ cb, float* __restrict__ xsf,
                            u16* __restrict__ xsb, size_t total) {
  size_t i = (size_t)blockIdx.x * blockDim.x + threadIdx.x;
  if (i >= total) return;                  // total = M * DI
  int c = (int)(i % DIi);
  size_t m = i / DIi;
  int t = (int)(m % Tt);
  size_t b = m / Tt;
  const float* w = cw + (size_t)c * DCc;
  float acc = 0.f;
#pragma unroll
  for (int j = 0; j < DCc; ++j) {
    int tt = t + j - (DCc - 1);
    if (tt >= 0) acc += w[j] * xz[((size_t)b * Tt + tt) * (2 * DIi) + c];
  }
  acc += cb[c];
  float v = acc * sigmoidf_(acc);
  xsf[i] = v;
  xsb[i] = f2bf(v);
}

// ---------------------------------------------------------------------------
// Selective scan: thread per (b, channel); DS=16 state in registers.
// Reads dt_lin (bias already added by GEMM), applies softplus; y gated by
// silu(z) and D skip; writes y as bf16 for out_proj GEMM.
// ---------------------------------------------------------------------------
__global__ void scan_k(const float* __restrict__ dtlin, const float* __restrict__ xs,
                       const float* __restrict__ xz,   const float* __restrict__ dbl,
                       const float* __restrict__ A_log, const float* __restrict__ Dp,
                       u16* __restrict__ ybf) {
  int idx = blockIdx.x * blockDim.x + threadIdx.x;   // 0 .. B*DI-1
  int c = idx % DIi;
  int b = idx / DIi;
  float a[DSs];
#pragma unroll
  for (int s = 0; s < DSs; ++s) a[s] = -__expf(A_log[(size_t)c * DSs + s]);
  const float Dv = Dp[c];
  float h[DSs] = {};
  for (int t = 0; t < Tt; ++t) {
    size_t m = (size_t)b * Tt + t;
    float dtv = dtlin[m * DIi + c];
    dtv = (dtv > 20.f) ? dtv : log1pf(__expf(dtv));   // softplus
    float xv = xs[m * DIi + c];
    float zv = xz[m * (2 * DIi) + DIi + c];
    const float* bm = dbl + m * 64 + Rr;              // B part
    const float* cm = dbl + m * 64 + Rr + DSs;        // C part
    float y = 0.f;
#pragma unroll
    for (int s = 0; s < DSs; ++s) {
      float dA = __expf(dtv * a[s]);
      h[s] = dA * h[s] + dtv * bm[s] * xv;
      y += h[s] * cm[s];
    }
    y = (y + xv * Dv) * (zv * sigmoidf_(zv));
    ybf[m * DIi + c] = f2bf(y);
  }
}

// ---------------------------------------------------------------------------
// Host-side orchestration
// ---------------------------------------------------------------------------
static inline size_t cdiv(size_t a, size_t b) { return (a + b - 1) / b; }

extern "C" void kernel_launch(void* const* d_in, const int* in_sizes, int n_in,
                              void* d_out, int out_size, void* d_ws, size_t ws_size,
                              hipStream_t stream) {
  (void)in_sizes; (void)n_in; (void)out_size;
  const int*   ids    = (const int*)  d_in[0];
  const float* tok    = (const float*)d_in[1];
  const float* pos    = (const float*)d_in[2];
  const float* ln1_g  = (const float*)d_in[3];
  const float* ln1_b  = (const float*)d_in[4];
  const float* inw    = (const float*)d_in[5];
  const float* cw     = (const float*)d_in[6];
  const float* cb     = (const float*)d_in[7];
  const float* xpw    = (const float*)d_in[8];
  const float* dtw    = (const float*)d_in[9];
  const float* dtb    = (const float*)d_in[10];
  const float* alog   = (const float*)d_in[11];
  const float* dprm   = (const float*)d_in[12];
  const float* outw   = (const float*)d_in[13];
  const float* ln2_g  = (const float*)d_in[14];
  const float* ln2_b  = (const float*)d_in[15];
  const float* fw1    = (const float*)d_in[16];
  const float* fb1    = (const float*)d_in[17];
  const float* fw2    = (const float*)d_in[18];
  const float* fb2    = (const float*)d_in[19];
  const float* lmw    = (const float*)d_in[20];
  const float* lmb    = (const float*)d_in[21];

  // Workspace carve-up (all 256B aligned by construction)
  char* ws = (char*)d_ws;
  size_t off = 0;
  auto carve = [&](size_t bytes) { char* p = ws + off; off += (bytes + 255) & ~(size_t)255; return p; };
  float* X    = (float*)carve((size_t)Mm * Dd * 4);        //  8 MB residual stream
  u16*   STG  = (u16*)  carve((size_t)Mm * 2048 * 2);      // 16 MB bf16 GEMM-input staging
  float* G0   = (float*)carve((size_t)Mm * 2048 * 4);      // 32 MB (xz / ffn hidden)
  float* G1   = (float*)carve((size_t)Mm * DIi * 4);       // 16 MB (dt_lin / proj tmp)
  float* XS   = (float*)carve((size_t)Mm * DIi * 4);       // 16 MB conv output f32
  float* DBL  = (float*)carve((size_t)Mm * 64 * 4);        //  1 MB x_proj output
  u16*   WB   = (u16*)  carve((size_t)2048 * 512 * 2);     //  2 MB bf16 weight scratch
  if (ws_size < off) return;   // insufficient scratch -> no-op (deterministic)

  const int TPB = 256;
  auto g1d = [&](size_t n) { return dim3((unsigned)cdiv(n, TPB)); };

  auto cast_w = [&](const float* w, size_t n) {
    castw_k<<<g1d(n), TPB, 0, stream>>>(w, WB, n);
  };
  auto gemm = [&](const u16* A, float* C, const float* bias, int N, int K) {
    dim3 grid(N / 64, Mm / 256);
    gemm_bf16_wmma<<<grid, 256, 0, stream>>>(A, WB, C, bias, N, K);
  };

  // Embedding
  embed_k<<<g1d((size_t)Mm * Dd), TPB, 0, stream>>>(ids, tok, pos, X, (size_t)Mm * Dd);

  for (int i = 0; i < Ll; ++i) {
    // ---- Mamba block ----
    ln_cast_k<<<Mm / 8, 256, 0, stream>>>(X, ln1_g + i * Dd, ln1_b + i * Dd, STG);

    cast_w(inw + (size_t)i * 2 * DIi * Dd, (size_t)2 * DIi * Dd);
    gemm(STG, G0, nullptr, 2 * DIi, Dd);                                // xz [M, 2048]

    conv_silu_k<<<g1d((size_t)Mm * DIi), TPB, 0, stream>>>(
        G0, cw + (size_t)i * DIi * DCc, cb + (size_t)i * DIi, XS, STG, (size_t)Mm * DIi);

    cast_w(xpw + (size_t)i * 64 * DIi, (size_t)64 * DIi);
    gemm(STG, DBL, nullptr, 64, DIi);                                   // dbl [M, 64]

    cast_rows_k<<<g1d((size_t)Mm * Rr), TPB, 0, stream>>>(
        DBL, 64, STG, Rr, 0, (size_t)Mm * Rr);                          // dt input bf16 [M,32]
    cast_w(dtw + (size_t)i * DIi * Rr, (size_t)DIi * Rr);
    gemm(STG, G1, dtb + (size_t)i * DIi, DIi, Rr);                      // dt_lin [M,1024] (+bias)

    scan_k<<<(Bb * DIi) / 256, 256, 0, stream>>>(
        G1, XS, G0, DBL, alog + (size_t)i * DIi * DSs, dprm + (size_t)i * DIi, STG);

    cast_w(outw + (size_t)i * Dd * DIi, (size_t)Dd * DIi);
    gemm(STG, G1, nullptr, Dd, DIi);                                    // mamba out [M,512]
    add_k<<<g1d((size_t)Mm * Dd), TPB, 0, stream>>>(X, G1, (size_t)Mm * Dd);

    // ---- FFN block ----
    ln_cast_k<<<Mm / 8, 256, 0, stream>>>(X, ln2_g + i * Dd, ln2_b + i * Dd, STG);

    cast_w(fw1 + (size_t)i * 4 * Dd * Dd, (size_t)4 * Dd * Dd);
    gemm(STG, G0, fb1 + (size_t)i * 4 * Dd, 4 * Dd, Dd);                // h [M,2048] (+b1)

    cast_rows_k<<<g1d((size_t)Mm * 4 * Dd), TPB, 0, stream>>>(
        G0, 4 * Dd, STG, 4 * Dd, 1, (size_t)Mm * 4 * Dd);               // relu + bf16

    cast_w(fw2 + (size_t)i * Dd * 4 * Dd, (size_t)Dd * 4 * Dd);
    gemm(STG, G1, fb2 + (size_t)i * Dd, Dd, 4 * Dd);                    // [M,512] (+b2)
    add_k<<<g1d((size_t)Mm * Dd), TPB, 0, stream>>>(X, G1, (size_t)Mm * Dd);
  }

  // ---- LM head ----
  cast_rows_k<<<g1d((size_t)Mm * Dd), TPB, 0, stream>>>(
      X, Dd, STG, Dd, 0, (size_t)Mm * Dd);
  cast_w(lmw, (size_t)Vv * Dd);
  gemm(STG, (float*)d_out, lmb, Vv, Dd);                                // logits [M,1024]
}
